// AttnDecoder_29274497089880
// MI455X (gfx1250) — matrix operand order repaired and checked by reading, not compiled, not run on hardware
//
#include <hip/hip_runtime.h>
#include <hip/hip_bf16.h>
#include <math.h>

// Problem constants (match reference)
#define BB 32
#define TT 64
#define SS 400
#define HH 512
#define EE 1024
#define EMBD 256
#define VV 32000
#define G4H (4 * HH)

typedef __attribute__((ext_vector_type(16))) __bf16 v16bf;
typedef __attribute__((ext_vector_type(8))) float v8f;

// Per-lane fragment loads (layouts per CDNA5 ISA 7.12.2, bf16 16x16x32):
//  A (16xK row-major): lane half h holds K {kb..kb+7} and {kb+16..kb+23}, kb=h*8
//  B (Kx16 from W[N,K] rows): lane half h holds K {kb..kb+7} and {kb+16..kb+23}, kb=h*8
__device__ __forceinline__ v16bf load_a_frag(const __bf16* p) {
  union { v16bf v; uint4 u[2]; } a;
  a.u[0] = *(const uint4*)(p);
  a.u[1] = *(const uint4*)(p + 16);
  return a.v;
}
__device__ __forceinline__ v16bf load_b_frag(const __bf16* p) {
  union { v16bf v; uint4 u[2]; } b;
  b.u[0] = *(const uint4*)(p);
  b.u[1] = *(const uint4*)(p + 16);
  return b.v;
}

// ---------------------------------------------------------------------------
// Register-blocked WMMA bf16 GEMM: C[M,N] = A[M,K] @ Wt[N,K]^T + bias[N]
// Each wave computes a 32x64 tile (2 M-tiles x 4 N-tiles, 8 accumulators):
// 8 WMMAs per 12 b128 loads. 4 waves/block -> block tile 32x256.
// __launch_bounds__(128, 1): full VGPR budget, no accumulator spills.
// Inner loop keeps one B fragment live at a time (~100 VGPR working set).
// Requires: M % 32 == 0, N % 256 == 0 per grid sizing below, K % 32 == 0.
// If swizzle != 0: row m = t*Bdim + b is written to C[(b*Tdim + t)*N + n].
// ---------------------------------------------------------------------------
__global__ void __launch_bounds__(128, 1)
gemm_xwt_bf16(const __bf16* __restrict__ A,
              const __bf16* __restrict__ Wt,
              const float* __restrict__ bias,
              float* __restrict__ C,
              int M, int N, int K,
              int swizzle, int Tdim, int Bdim) {
  const int lane = threadIdx.x & 31;
  const int wave = threadIdx.x >> 5;
  const int n0 = (blockIdx.x * (blockDim.x >> 5) + wave) * 64;
  const int m0 = blockIdx.y * 32;
  if (n0 >= N || m0 >= M) return;  // uniform per wave -> EXEC stays all-ones

  const int half = lane >> 4;  // 0: lanes 0-15, 1: lanes 16-31
  const int l16 = lane & 15;

  // Per-lane base pointers for the 2 A row-tiles and 4 B col-tiles
  const __bf16* Ap0 = A + (size_t)(m0 + l16) * K + half * 8;
  const __bf16* Ap1 = A + (size_t)(m0 + 16 + l16) * K + half * 8;
  const __bf16* Bp = Wt + (size_t)(n0 + l16) * K + half * 8;
  const size_t bstride = (size_t)16 * K;  // between consecutive N tiles

  v8f acc[2][4] = {};
  for (int k = 0; k < K; k += 32) {
    v16bf a0 = load_a_frag(Ap0 + k);
    v16bf a1 = load_a_frag(Ap1 + k);
#pragma unroll
    for (int j = 0; j < 4; ++j) {
      v16bf bf = load_b_frag(Bp + (size_t)j * bstride + k);
      acc[0][j] = __builtin_amdgcn_wmma_f32_16x16x32_bf16(
          false, a0, false, bf, (short)0, acc[0][j], false, false);
      acc[1][j] = __builtin_amdgcn_wmma_f32_16x16x32_bf16(
          false, a1, false, bf, (short)0, acc[1][j], false, false);
    }
  }

  // C/D layout per tile: lane column = l16 ; VGPR r holds row r + half*8
#pragma unroll
  for (int j = 0; j < 4; ++j) {
    const int n = n0 + j * 16 + l16;
    const float bv = bias ? bias[n] : 0.0f;
#pragma unroll
    for (int i = 0; i < 2; ++i) {
#pragma unroll
      for (int r = 0; r < 8; ++r) {
        const int m = m0 + i * 16 + r + half * 8;
        size_t o;
        if (swizzle) {
          const int bb = m % Bdim;
          const int tt = m / Bdim;
          o = ((size_t)bb * Tdim + tt) * N + n;
        } else {
          o = (size_t)m * N + n;
        }
        C[o] = acc[i][j][r] + bv;
      }
    }
  }
}

// ---------------------------------------------------------------------------
// Small helper kernels
// ---------------------------------------------------------------------------
__global__ void f32_to_bf16_kernel(const float* __restrict__ in,
                                   __bf16* __restrict__ out, size_t n) {
  size_t i = (size_t)blockIdx.x * blockDim.x + threadIdx.x;
  if (i < n) out[i] = (__bf16)in[i];
}

__global__ void embed_gather_kernel(const int* __restrict__ target,
                                    const float* __restrict__ emb,
                                    __bf16* __restrict__ x_all) {
  // x_all[(t*B + b)*EMB + e] = emb[target[b*T + t]*EMB + e]
  size_t i = (size_t)blockIdx.x * blockDim.x + threadIdx.x;
  const size_t n = (size_t)TT * BB * EMBD;
  if (i >= n) return;
  int e = (int)(i % EMBD);
  int tb = (int)(i / EMBD);
  int b = tb % BB;
  int t = tb / BB;
  int tok = target[b * TT + t];
  x_all[i] = (__bf16)emb[(size_t)tok * EMBD + e];
}

__global__ void bias_comb_kernel(const float* __restrict__ b0,
                                 const float* __restrict__ b1,
                                 float* __restrict__ out) {
  int i = blockIdx.x * blockDim.x + threadIdx.x;
  if (i < G4H) out[i] = b0[i] + b1[i];
}

__global__ void init_state_kernel(const float* __restrict__ h0,
                                  const float* __restrict__ c0,
                                  float* __restrict__ h, float* __restrict__ c,
                                  __bf16* __restrict__ h_bf) {
  int i = blockIdx.x * blockDim.x + threadIdx.x;
  if (i < BB * HH) {
    float hv = h0[i];
    h[i] = hv;
    c[i] = c0[i];
    h_bf[i] = (__bf16)hv;
  }
}

__device__ __forceinline__ float sigf(float x) { return 1.0f / (1.0f + expf(-x)); }

__global__ void lstm_update_kernel(const float* __restrict__ gx,  // [B,4H] (t offset applied)
                                   const float* __restrict__ gh,  // [B,4H]
                                   float* __restrict__ c, float* __restrict__ h,
                                   __bf16* __restrict__ h_bf) {
  int idx = blockIdx.x * blockDim.x + threadIdx.x;
  if (idx >= BB * HH) return;
  int b = idx / HH, j = idx % HH;
  const float* g0 = gx + (size_t)b * G4H;
  const float* g1 = gh + (size_t)b * G4H;
  float gi = g0[j] + g1[j];
  float gf = g0[HH + j] + g1[HH + j];
  float gg = g0[2 * HH + j] + g1[2 * HH + j];
  float go = g0[3 * HH + j] + g1[3 * HH + j];
  float cc = sigf(gf) * c[idx] + sigf(gi) * tanhf(gg);
  float hh = sigf(go) * tanhf(cc);
  c[idx] = cc;
  h[idx] = hh;
  h_bf[idx] = (__bf16)hh;
}

__device__ __forceinline__ float block_reduce_max(float v, float* red) {
  __syncthreads();
  int lane = threadIdx.x & 31, wid = threadIdx.x >> 5;
  for (int o = 16; o > 0; o >>= 1) v = fmaxf(v, __shfl_down(v, o, 32));
  if (lane == 0) red[wid] = v;
  __syncthreads();
  if (wid == 0) {
    int nw = blockDim.x >> 5;
    float x = (lane < nw) ? red[lane] : -INFINITY;
    for (int o = 16; o > 0; o >>= 1) x = fmaxf(x, __shfl_down(x, o, 32));
    if (lane == 0) red[0] = x;
  }
  __syncthreads();
  return red[0];
}

__device__ __forceinline__ float block_reduce_sum(float v, float* red) {
  __syncthreads();
  int lane = threadIdx.x & 31, wid = threadIdx.x >> 5;
  for (int o = 16; o > 0; o >>= 1) v += __shfl_down(v, o, 32);
  if (lane == 0) red[wid] = v;
  __syncthreads();
  if (wid == 0) {
    int nw = blockDim.x >> 5;
    float x = (lane < nw) ? red[lane] : 0.0f;
    for (int o = 16; o > 0; o >>= 1) x += __shfl_down(x, o, 32);
    if (lane == 0) red[0] = x;
  }
  __syncthreads();
  return red[0];
}

// One block per batch element b. scores -> softmax -> ctx.
__global__ void attention_kernel(const float* __restrict__ h,
                                 const float* __restrict__ keys,  // [S*B, H]
                                 const float* __restrict__ enc,   // [S*B, E]
                                 const int* __restrict__ attn_mask,
                                 float* __restrict__ ctx) {       // [B, E]
  __shared__ float hs[HH];
  __shared__ float sc[SS];
  __shared__ float red[32];
  const int b = blockIdx.x;
  const int tid = threadIdx.x;
  const int nt = blockDim.x;

  for (int k = tid; k < HH; k += nt) hs[k] = h[b * HH + k];
  __syncthreads();

  for (int s = tid; s < SS; s += nt) {
    const float* kr = keys + ((size_t)s * BB + b) * HH;
    float acc = 0.0f;
    for (int k = 0; k < HH; k += 4) {
      acc += hs[k] * kr[k] + hs[k + 1] * kr[k + 1] +
             hs[k + 2] * kr[k + 2] + hs[k + 3] * kr[k + 3];
    }
    sc[s] = (attn_mask[b * SS + s] == 0) ? -1e18f : acc;
  }

  float mx = -INFINITY;
  for (int s = tid; s < SS; s += nt) mx = fmaxf(mx, sc[s]);
  mx = block_reduce_max(mx, red);

  float sum = 0.0f;
  for (int s = tid; s < SS; s += nt) {
    float e = expf(sc[s] - mx);
    sc[s] = e;
    sum += e;
  }
  sum = block_reduce_sum(sum, red);
  const float inv = 1.0f / sum;
  __syncthreads();
  for (int s = tid; s < SS; s += nt) sc[s] *= inv;
  __syncthreads();

  for (int e = tid; e < EE; e += nt) {
    float acc = 0.0f;
    for (int s = 0; s < SS; ++s) acc += sc[s] * enc[((size_t)s * BB + b) * EE + e];
    ctx[(size_t)b * EE + e] = acc;
  }
}

// cat[b,j] = tanh(b_concat[j] + ctx[b,:]·Wc[j,0:E] + h[b,:]·Wc[j,E:E+H])
__global__ void concat_proj_kernel(const float* __restrict__ ctx,
                                   const float* __restrict__ h,
                                   const float* __restrict__ Wc,  // [H, E+H]
                                   const float* __restrict__ bc,
                                   __bf16* __restrict__ cat_bf) { // [B, H] (t offset applied)
  int idx = blockIdx.x * blockDim.x + threadIdx.x;
  if (idx >= BB * HH) return;
  int b = idx / HH, j = idx % HH;
  const float* w = Wc + (size_t)j * (EE + HH);
  const float* cb = ctx + (size_t)b * EE;
  const float* hb = h + (size_t)b * HH;
  float acc = bc[j];
  for (int k = 0; k < EE; ++k) acc += cb[k] * w[k];
  for (int k = 0; k < HH; ++k) acc += hb[k] * w[EE + k];
  cat_bf[idx] = (__bf16)tanhf(acc);
}

// In-place log_softmax over last dim V; one block per (b,t) row.
__global__ void log_softmax_kernel(float* __restrict__ out) {
  __shared__ float red[32];
  float* p = out + (size_t)blockIdx.x * VV;
  const int tid = threadIdx.x, nt = blockDim.x;

  float mx = -INFINITY;
  for (int v = tid; v < VV; v += nt) mx = fmaxf(mx, p[v]);
  mx = block_reduce_max(mx, red);

  float sum = 0.0f;
  for (int v = tid; v < VV; v += nt) sum += expf(p[v] - mx);
  sum = block_reduce_sum(sum, red);
  const float lse = mx + logf(sum);

  for (int v = tid; v < VV; v += nt) p[v] = p[v] - lse;
}

// ---------------------------------------------------------------------------
// Host launch
// ---------------------------------------------------------------------------
static inline size_t align_up(size_t x, size_t a) { return (x + a - 1) & ~(a - 1); }

extern "C" void kernel_launch(void* const* d_in, const int* in_sizes, int n_in,
                              void* d_out, int out_size, void* d_ws, size_t ws_size,
                              hipStream_t stream) {
  (void)in_sizes; (void)n_in; (void)out_size; (void)ws_size;

  const int*   target   = (const int*)d_in[0];
  const float* h0       = (const float*)d_in[1];
  const float* c0       = (const float*)d_in[2];
  const float* enc_outs = (const float*)d_in[3];
  const int*   attnmask = (const int*)d_in[4];
  const float* emb      = (const float*)d_in[5];
  const float* W_ih     = (const float*)d_in[6];
  const float* b_ih     = (const float*)d_in[7];
  const float* W_hh     = (const float*)d_in[8];
  const float* b_hh     = (const float*)d_in[9];
  const float* W_attn   = (const float*)d_in[10];
  const float* b_attn   = (const float*)d_in[11];
  const float* W_concat = (const float*)d_in[12];
  const float* b_concat = (const float*)d_in[13];
  const float* W_out    = (const float*)d_in[14];
  const float* b_out    = (const float*)d_in[15];
  float* out = (float*)d_out;

  // Workspace carve-out
  char* ws = (char*)d_ws;
  size_t cur = 0;
  auto carve = [&](size_t bytes) {
    void* p = ws + cur;
    cur = align_up(cur + bytes, 256);
    return p;
  };
  __bf16* enc_bf   = (__bf16*)carve((size_t)SS * BB * EE * 2);
  __bf16* Wattn_bf = (__bf16*)carve((size_t)HH * EE * 2);
  __bf16* Wih_bf   = (__bf16*)carve((size_t)G4H * EMBD * 2);
  __bf16* Whh_bf   = (__bf16*)carve((size_t)G4H * HH * 2);
  __bf16* Wout_bf  = (__bf16*)carve((size_t)VV * HH * 2);
  __bf16* x_all_bf = (__bf16*)carve((size_t)TT * BB * EMBD * 2);
  __bf16* h_bf     = (__bf16*)carve((size_t)BB * HH * 2);
  __bf16* cat_bf   = (__bf16*)carve((size_t)TT * BB * HH * 2);
  float*  biasc    = (float*)carve((size_t)G4H * 4);
  float*  gates_x  = (float*)carve((size_t)TT * BB * G4H * 4);
  float*  keys     = (float*)carve((size_t)SS * BB * HH * 4);
  float*  gatesh   = (float*)carve((size_t)BB * G4H * 4);
  float*  hbuf     = (float*)carve((size_t)BB * HH * 4);
  float*  cbuf     = (float*)carve((size_t)BB * HH * 4);
  float*  ctx      = (float*)carve((size_t)BB * EE * 4);

  const int THR = 256;
  auto blocks = [](size_t n, int t) { return (unsigned)((n + t - 1) / t); };

  // 1) fp32 -> bf16 conversions (weights + encoder outputs)
  f32_to_bf16_kernel<<<blocks((size_t)SS * BB * EE, THR), THR, 0, stream>>>(enc_outs, enc_bf, (size_t)SS * BB * EE);
  f32_to_bf16_kernel<<<blocks((size_t)HH * EE, THR), THR, 0, stream>>>(W_attn, Wattn_bf, (size_t)HH * EE);
  f32_to_bf16_kernel<<<blocks((size_t)G4H * EMBD, THR), THR, 0, stream>>>(W_ih, Wih_bf, (size_t)G4H * EMBD);
  f32_to_bf16_kernel<<<blocks((size_t)G4H * HH, THR), THR, 0, stream>>>(W_hh, Whh_bf, (size_t)G4H * HH);
  f32_to_bf16_kernel<<<blocks((size_t)VV * HH, THR), THR, 0, stream>>>(W_out, Wout_bf, (size_t)VV * HH);

  // 2) embedding gather (to bf16), combined LSTM bias, initial state
  embed_gather_kernel<<<blocks((size_t)TT * BB * EMBD, THR), THR, 0, stream>>>(target, emb, x_all_bf);
  bias_comb_kernel<<<blocks(G4H, THR), THR, 0, stream>>>(b_ih, b_hh, biasc);
  init_state_kernel<<<blocks(BB * HH, THR), THR, 0, stream>>>(h0, c0, hbuf, cbuf, h_bf);

  // Wave tile = 32x64, block (128 thr) tile = 32x256.
  // 3) keys_proj = enc @ W_attn^T + b_attn : [S*B=12800, E=1024] x [H=512]
  {
    dim3 g(HH / 256, SS * BB / 32);  // (2, 400)
    gemm_xwt_bf16<<<g, 128, 0, stream>>>(enc_bf, Wattn_bf, b_attn, keys,
                                         SS * BB, HH, EE, 0, 0, 0);
  }
  // 4) gates_x = x_all @ W_ih^T + (b_ih + b_hh) : [T*B=2048, EMB=256] x [4H=2048]
  {
    dim3 g(G4H / 256, TT * BB / 32);  // (8, 64)
    gemm_xwt_bf16<<<g, 128, 0, stream>>>(x_all_bf, Wih_bf, biasc, gates_x,
                                         TT * BB, G4H, EMBD, 0, 0, 0);
  }

  // 5) sequential scan over T
  for (int t = 0; t < TT; ++t) {
    // gatesh = h @ W_hh^T : [32, 512] x [2048]
    dim3 gh(G4H / 256, BB / 32);  // (8, 1)
    gemm_xwt_bf16<<<gh, 128, 0, stream>>>(h_bf, Whh_bf, nullptr, gatesh,
                                          BB, G4H, HH, 0, 0, 0);
    lstm_update_kernel<<<blocks(BB * HH, THR), THR, 0, stream>>>(
        gates_x + (size_t)t * BB * G4H, gatesh, cbuf, hbuf, h_bf);
    attention_kernel<<<BB, THR, 0, stream>>>(hbuf, keys, enc_outs, attnmask, ctx);
    concat_proj_kernel<<<blocks(BB * HH, THR), THR, 0, stream>>>(
        ctx, hbuf, W_concat, b_concat, cat_bf + (size_t)t * BB * HH);
  }

  // 6) logits = cat_all @ W_out^T + b_out : [T*B=2048, H=512] x [V=32000]
  //    rows m = t*B + b written directly to d_out[(b*T + t)*V + :]
  {
    dim3 g(VV / 256, TT * BB / 32);  // (125, 64)
    gemm_xwt_bf16<<<g, 128, 0, stream>>>(cat_bf, Wout_bf, b_out, out,
                                         TT * BB, VV, HH, 1, TT, BB);
  }
  // 7) in-place log_softmax over V per (b,t) row
  log_softmax_kernel<<<TT * BB, 512, 0, stream>>>(out);
}
